// SelfAttentionPairDim1_44444321579150
// MI455X (gfx1250) — compile-verified, matrix operand order checked by reading, not run on hardware
//
#include <hip/hip_runtime.h>

typedef __attribute__((ext_vector_type(16))) __bf16 bf16x16;
typedef __attribute__((ext_vector_type(8)))  float  f32x8;

#define DD 128
#define NN 256
#define LN_EPS 1e-5f

// LDS row strides (elements); all chosen so 16-byte alignment holds everywhere.
#define SSTR 136   // S/Q/K rows: 272 B
#define VSTR 264   // Vt rows: 528 B
#define PSTR 136   // per-wave scratch rows
#define OSTR 132   // per-wave fp32 out scratch rows (aliases S region)

// Native bf16 conversion (gfx1250 has hardware f32->bf16 converts).
__device__ __forceinline__ unsigned short f2bf(float f) {
  union { __bf16 b; unsigned short u; } x;
  x.b = (__bf16)f;
  return x.u;
}

// Packed pair: lets clang fuse into v_cvt_pk_bf16_f32.
__device__ __forceinline__ unsigned int pk2bf(float lo, float hi) {
  union { __bf16 b[2]; unsigned int u; } x;
  x.b[0] = (__bf16)lo;
  x.b[1] = (__bf16)hi;
  return x.u;
}

__device__ __forceinline__ f32x8 wmma32(bf16x16 a, bf16x16 b, f32x8 c) {
  return __builtin_amdgcn_wmma_f32_16x16x32_bf16(false, a, false, b, (short)0, c,
                                                 false, false);
}

union FragU { bf16x16 f; uint4 q[2]; };

// Load one 16x32 bf16 operand fragment. p = row_base + k0 + 8*hi (elements).
// Per ISA layout: lanes<16 hold K {k0..k0+7, k0+16..k0+23}; lanes>=16 are +8.
__device__ __forceinline__ bf16x16 load_frag(const unsigned short* p) {
  FragU u;
  u.q[0] = *(const uint4*)(p);
  u.q[1] = *(const uint4*)(p + 16);
  return u.f;
}

__global__ __launch_bounds__(256) void prep_weights(
    const float* __restrict__ Wq, const float* __restrict__ Wk,
    const float* __restrict__ Wv, const float* __restrict__ Wo,
    unsigned short* __restrict__ wt) {
  int idx = blockIdx.x * 256 + threadIdx.x;   // 0..16383
  #pragma unroll
  for (int e = 0; e < 4; ++e) {
    int o   = idx * 4 + e;                    // 0..65535
    int mat = o >> 14;
    int rem = o & 16383;
    int n   = rem >> 7;
    int k   = rem & 127;
    const float* W = (mat == 0) ? Wq : (mat == 1) ? Wk : (mat == 2) ? Wv : Wo;
    wt[o] = f2bf(W[k * DD + n]);              // store W^T in bf16
  }
}

__global__ __launch_bounds__(256) void pair_attn(
    const float* __restrict__ S,              // [B*N, N, D] fp32
    const unsigned short* __restrict__ wt,    // [4][D][D] bf16, transposed
    const float* __restrict__ bq, const float* __restrict__ bk,
    const float* __restrict__ bv, const float* __restrict__ bo,
    const float* __restrict__ ln_g, const float* __restrict__ ln_b,
    float* __restrict__ out) {
  __shared__ unsigned short Ssh[NN * SSTR];   // S bf16; later fp32 LN scratch
  __shared__ unsigned short Qsh[NN * SSTR];   // Q row-major bf16
  __shared__ unsigned short Ksh[NN * SSTR];   // K row-major bf16
  __shared__ unsigned short Vt [DD * VSTR];   // V transposed bf16
  __shared__ unsigned short Pscr[8 * 16 * PSTR]; // per-wave probs/ctx scratch

  const int tid  = threadIdx.x;
  const int lane = tid & 31;
  const int w    = tid >> 5;
  const int lr   = lane & 15;
  const int hi   = lane >> 4;
  const long long base = (long long)blockIdx.x * (NN * DD);
  const float* Sg = S + base;

  // Warm L2 with the (shared, tiny) bf16 weight buffer.
  __builtin_prefetch(wt + w * 8192, 0, 1);

  // ---- Phase 0: load S tile (fp32) -> LDS bf16, one row per thread ----
  {
    const float* row = Sg + tid * DD;
    unsigned short* dst = Ssh + tid * SSTR;
    for (int c = 0; c < DD; c += 8) {
      float4 a = *(const float4*)(row + c);
      float4 b = *(const float4*)(row + c + 4);
      uint4 pk;
      pk.x = pk2bf(a.x, a.y);
      pk.y = pk2bf(a.z, a.w);
      pk.z = pk2bf(b.x, b.y);
      pk.w = pk2bf(b.z, b.w);
      *(uint4*)(dst + c) = pk;
    }
  }
  __syncthreads();

  // ---- Phase 1: Q = S*Wq+bq, K = S*Wk+bk, V^T = (S*Wv+bv)^T ----
  for (int job = w; job < 384; job += 8) {
    int mat = job >> 7;            // 0:Q 1:K 2:V
    int rem = job & 127;
    int mb  = rem >> 3;            // row block 0..15
    int nb  = rem & 7;             // col block 0..7
    const float* bp = (mat == 0) ? bq : (mat == 1) ? bk : bv;
    float bias = bp[nb * 16 + lr];
    f32x8 c;
    #pragma unroll
    for (int r = 0; r < 8; ++r) c[r] = bias;
    const unsigned short* arow = Ssh + (mb * 16 + lr) * SSTR + 8 * hi;
    const unsigned short* brow = wt + mat * (DD * DD) + (nb * 16 + lr) * DD + 8 * hi;
    #pragma unroll
    for (int kk = 0; kk < 4; ++kk) {
      bf16x16 a = load_frag(arow + kk * 32);
      bf16x16 b = load_frag(brow + kk * 32);
      c = wmma32(a, b, c);
    }
    if (mat == 2) {                // transposed store: contiguous in m
      unsigned short* dst = Vt + (nb * 16 + lr) * VSTR + mb * 16 + 8 * hi;
      uint4 pk;
      pk.x = pk2bf(c[0], c[1]);
      pk.y = pk2bf(c[2], c[3]);
      pk.z = pk2bf(c[4], c[5]);
      pk.w = pk2bf(c[6], c[7]);
      *(uint4*)dst = pk;
    } else {
      unsigned short* dst = (mat == 0) ? Qsh : Ksh;
      #pragma unroll
      for (int r = 0; r < 8; ++r)
        dst[(mb * 16 + r + 8 * hi) * SSTR + nb * 16 + lr] = f2bf(c[r]);
    }
  }
  __syncthreads();

  // ---- Phase 2: per-wave attention over 2 j-blocks of 16 rows ----
  const float scale = 0.08838834764831845f;   // 1/sqrt(128)
  unsigned short* pscr = Pscr + w * (16 * PSTR);
  float* outscr = (float*)Ssh + w * (16 * OSTR);  // S bf16 no longer needed
  const f32x8 vzero = {0.f, 0.f, 0.f, 0.f, 0.f, 0.f, 0.f, 0.f};

  for (int jb2 = 0; jb2 < 2; ++jb2) {
    const int J0 = (w * 2 + jb2) * 16;
    f32x8 cacc[8];
    #pragma unroll
    for (int nb = 0; nb < 8; ++nb) cacc[nb] = vzero;

    const unsigned short* krow = Ksh + (J0 + lr) * SSTR + 8 * hi;
    for (int kb = 0; kb < 8; ++kb) {          // 32 q-rows per chunk
      #pragma unroll
      for (int t = 0; t < 2; ++t) {
        f32x8 s = vzero;
        const unsigned short* qrow =
            Qsh + (kb * 32 + t * 16 + lr) * SSTR + 8 * hi;
        #pragma unroll
        for (int dk = 0; dk < 4; ++dk) {
          bf16x16 a = load_frag(krow + dk * 32);   // K rows (j)
          bf16x16 b = load_frag(qrow + dk * 32);   // Q^T columns
          s = wmma32(a, b, s);
        }
        #pragma unroll
        for (int r = 0; r < 8; ++r) {
          float v = fmaxf(s[r], 0.f) * scale;      // relu + 1/sqrt(D)
          pscr[(r + 8 * hi) * PSTR + t * 16 + lr] = f2bf(v);
        }
      }
      asm volatile("" ::: "memory");  // order LDS bounce (same-wave DS is in-order)
      bf16x16 ap = load_frag(pscr + lr * PSTR + 8 * hi);  // probs as A operand
      #pragma unroll
      for (int nb = 0; nb < 8; ++nb) {
        bf16x16 bvf = load_frag(Vt + (nb * 16 + lr) * VSTR + kb * 32 + 8 * hi);
        cacc[nb] = wmma32(ap, bvf, cacc[nb]);
      }
    }

    // ctx -> bf16 scratch (A operand for output projection)
    #pragma unroll
    for (int nb = 0; nb < 8; ++nb)
      #pragma unroll
      for (int r = 0; r < 8; ++r)
        pscr[(r + 8 * hi) * PSTR + nb * 16 + lr] = f2bf(cacc[nb][r]);
    asm volatile("" ::: "memory");

    // out = ctx*Wo + bo -> fp32 scratch
    #pragma unroll
    for (int nb = 0; nb < 8; ++nb) {
      float bias = bo[nb * 16 + lr];
      f32x8 o;
      #pragma unroll
      for (int r = 0; r < 8; ++r) o[r] = bias;
      const unsigned short* brow = wt + 3 * (DD * DD) + (nb * 16 + lr) * DD + 8 * hi;
      const unsigned short* arow = pscr + lr * PSTR + 8 * hi;
      #pragma unroll
      for (int dk = 0; dk < 4; ++dk) {
        bf16x16 a = load_frag(arow + dk * 32);
        bf16x16 b = load_frag(brow + dk * 32);
        o = wmma32(a, b, o);
      }
      #pragma unroll
      for (int r = 0; r < 8; ++r)
        outscr[(r + 8 * hi) * OSTR + nb * 16 + lr] = o[r];
    }
    asm volatile("" ::: "memory");

    // residual (fp32 S reload) + LayerNorm.
    // Full wave: 2 lanes per row, 64 columns each; combine via shfl_xor(16).
    {
      const int j = J0 + lr;
      const int c0 = hi * 64;                     // column half
      const float* srow = Sg + j * DD + c0;
      float* xrow = outscr + lr * OSTR + c0;
      float sum = 0.f, sq = 0.f;
      for (int d = 0; d < 64; d += 4) {
        float4 sv = *(const float4*)(srow + d);
        float4 cv = *(const float4*)(xrow + d);
        float x0 = cv.x + sv.x, x1 = cv.y + sv.y;
        float x2 = cv.z + sv.z, x3 = cv.w + sv.w;
        sum += x0 + x1 + x2 + x3;
        sq  += x0 * x0 + x1 * x1 + x2 * x2 + x3 * x3;
        float4 xw; xw.x = x0; xw.y = x1; xw.z = x2; xw.w = x3;
        *(float4*)(xrow + d) = xw;
      }
      sum += __shfl_xor(sum, 16, 32);             // combine the two halves
      sq  += __shfl_xor(sq, 16, 32);
      float mu  = sum * (1.f / 128.f);
      float var = sq * (1.f / 128.f) - mu * mu;
      float rs  = rsqrtf(var + LN_EPS);
      float* orow = out + base + (long long)j * DD + c0;
      for (int d = 0; d < 64; d += 4) {
        float4 xv = *(const float4*)(xrow + d);
        float4 g  = *(const float4*)(ln_g + c0 + d);
        float4 be = *(const float4*)(ln_b + c0 + d);
        float4 r;
        r.x = (xv.x - mu) * rs * g.x + be.x;
        r.y = (xv.y - mu) * rs * g.y + be.y;
        r.z = (xv.z - mu) * rs * g.z + be.z;
        r.w = (xv.w - mu) * rs * g.w + be.w;
        *(float4*)(orow + d) = r;
      }
    }
    asm volatile("" ::: "memory");
  }
}

extern "C" void kernel_launch(void* const* d_in, const int* in_sizes, int n_in,
                              void* d_out, int out_size, void* d_ws, size_t ws_size,
                              hipStream_t stream) {
  (void)in_sizes; (void)n_in; (void)out_size; (void)ws_size;
  const float* S  = (const float*)d_in[1];   // structure_matrix [B,N,N,D]
  const float* Wq = (const float*)d_in[2];
  const float* bq = (const float*)d_in[3];
  const float* Wk = (const float*)d_in[4];
  const float* bk = (const float*)d_in[5];
  const float* Wv = (const float*)d_in[6];
  const float* bv = (const float*)d_in[7];
  const float* Wo = (const float*)d_in[8];
  const float* bo = (const float*)d_in[9];
  const float* g  = (const float*)d_in[10];
  const float* be = (const float*)d_in[11];
  unsigned short* wt = (unsigned short*)d_ws;  // 4 x 128 x 128 bf16 (128 KB)

  prep_weights<<<64, 256, 0, stream>>>(Wq, Wk, Wv, Wo, wt);
  pair_attn<<<512, 256, 0, stream>>>(S, wt, bq, bk, bv, bo, g, be, (float*)d_out);
}
